// patching_STconv_prev_83408264888895
// MI455X (gfx1250) — matrix operand order repaired and checked by reading, not compiled, not run on hardware
//
#include <hip/hip_runtime.h>
#include <hip/hip_bf16.h>
#include <math.h>

// ---------------- problem constants ----------------
#define BB  8
#define NN  1024
#define TT  64
#define CC  16
#define OCC 16
#define EE  16384

// ---------------- WMMA plumbing (gfx1250, wave32) ----------------
typedef __attribute__((ext_vector_type(16))) __bf16 v16bf;
typedef __attribute__((ext_vector_type(8)))  float  v8f;

struct alignas(16) B16 { unsigned int w[4]; };   // 16-byte chunk

__device__ __forceinline__ v8f wmma_bf16(v16bf a, v16bf b, v8f c) {
  return __builtin_amdgcn_wmma_f32_16x16x32_bf16(
      false, a, false, b, (short)0, c, false, false);
}

// Build a v16bf fragment from two contiguous 8-element (16B) runs.
__device__ __forceinline__ v16bf ld_frag(const __bf16* p0, const __bf16* p1) {
  union { v16bf v; B16 q[2]; } u;
  u.q[0] = *(const B16*)p0;
  u.q[1] = *(const B16*)p1;
  return u.v;
}

// Branch-free tanh: 1 - 2/(1+e^{2x}) via v_exp_f32 + v_rcp_f32.
// Saturates correctly: x->+inf => exp2->inf => 1;  x->-inf => exp2->0 => -1.
__device__ __forceinline__ float fast_tanh(float x) {
  float t = __builtin_amdgcn_exp2f(x * 2.8853900817779268f);  // 2*log2(e)
  return 1.0f - 2.0f * __builtin_amdgcn_rcpf(t + 1.0f);
}

// ISA 7.12.2 16-bit A layout: lane<16: runs start K=0 and K=16;
//                             lane>=16: runs start K=8 and K=24.
// 16-bit B layout: same K runs as A (lane<16: K=0 and K=16; lane>=16: K=8 and K=24).

// ---------------- K0: degree -> dinv -------------------
__global__ __launch_bounds__(1024) void k_deg(const int* __restrict__ ei,
                                              float* __restrict__ dinv) {
  __shared__ float s_deg[NN];
  const int tid = threadIdx.x;
  s_deg[tid] = 0.f;
  __syncthreads();
  const int* col = ei + EE;
  for (int e = tid; e < EE; e += 1024) atomicAdd(&s_deg[col[e]], 1.0f);
  __syncthreads();
  dinv[tid] = rsqrtf(s_deg[tid] + 1.0f);
}

// ---------------- K1: fused 4-branch temporal conv (WMMA) ----------------
// One wave per (b,n). Padded X rows t=-6..69 in LDS. Weights transposed+zero
// padded per 32-K chunk: s_wT[chunk][oc][32].  A reads beyond Keff hit valid
// LDS rows and multiply zero weight rows -> exact zeros, no predication.
// Weight (B) fragments depend only on the K-chunk: hoisted out of the m loop.
template<int L, int CB, int NCH, int BR>
__device__ __forceinline__ void tconv_branch(const __bf16* sx, const __bf16* swT,
                                             const float* bias_p, float* op,
                                             int lane) {
  const int oc = lane & 15;
  const int half = lane >> 4;            // 0: lanes 0-15, 1: lanes 16-31
  const float bias = bias_p[oc];

  v16bf bw[NCH];                         // hoisted weight fragments
#pragma unroll
  for (int ch = 0; ch < NCH; ++ch) {
    const __bf16* pb = swT + ((CB + ch) * 16 + oc) * 32 + (half ? 8 : 0);
    bw[ch] = ld_frag(pb, pb + 16);
  }

#pragma unroll
  for (int m = 0; m < 4; ++m) {
    const int t = m * 16 + oc;           // M row = t
    v8f acc = {};
#pragma unroll
    for (int ch = 0; ch < NCH; ++ch) {
      // A: two 16B runs, rows tap=2ch and 2ch+1, col 0 or 8
      const __bf16* pa0 = sx + (t - L + 2 * ch + 6) * 16 + (half ? 8 : 0);
      const __bf16* pa1 = pa0 + 16;      // next padded row
      v16bf a = ld_frag(pa0, pa1);
      acc = wmma_bf16(a, bw[ch], acc);
    }
#pragma unroll
    for (int r = 0; r < 8; ++r) {
      int tt = m * 16 + r + (half ? 8 : 0);
      op[tt * 80 + BR * 16 + oc] = fast_tanh(acc[r] + bias);
    }
  }
}

__global__ __launch_bounds__(256) void k_tconv(
    const float* __restrict__ x,
    const float* __restrict__ w2,  const float* __restrict__ bb2,
    const float* __restrict__ w3,  const float* __restrict__ bb3,
    const float* __restrict__ w6,  const float* __restrict__ bb6,
    const float* __restrict__ w12, const float* __restrict__ bb12,
    float* __restrict__ out) {
  __shared__ __align__(16) __bf16 s_wT[12 * 16 * 32]; // [chunk][oc][k], 12KB
  __shared__ __align__(16) __bf16 s_x[8][76 * 16];    // per-wave padded tile
  const int tid  = threadIdx.x;
  const int lane = tid & 31;
  const int wv   = tid >> 5;

  for (int i = tid; i < 12 * 16 * 32; i += 256) s_wT[i] = (__bf16)0.f;
  __syncthreads();

  const float* ws_[4] = {w2, w3, w6, w12};
  const int ks_[4] = {2, 3, 6, 12};
  const int cb_[4] = {0, 1, 3, 6};
  for (int br = 0; br < 4; ++br) {
    const int ks = ks_[br];
    const int tot = OCC * CC * ks;
    const float* w = ws_[br];
    for (int i = tid; i < tot; i += 256) {
      int oc = i / (CC * ks), rem = i % (CC * ks);
      int c = rem / ks, tap = rem % ks;
      int kk = tap * 16 + c;                       // K index = (tap, c)
      s_wT[(cb_[br] + (kk >> 5)) * 512 + oc * 32 + (kk & 31)] = (__bf16)w[i];
    }
  }

  const long bn = (long)blockIdx.x * 8 + wv;       // flat b*N+n
  const float* xp = x + bn * (TT * CC);
  for (int i = lane; i < 76 * 16; i += 32) {       // edge-replication pad
    int r = i >> 4, c = i & 15;
    int t = r - 6; t = t < 0 ? 0 : (t > 63 ? 63 : t);
    s_x[wv][i] = (__bf16)xp[t * CC + c];
  }
  __syncthreads();

  float* op = out + bn * (TT * 80);
  const __bf16* sx = s_x[wv];
  tconv_branch<0, 0, 1, 0>(sx, s_wT, bb2,  op, lane);
  tconv_branch<1, 1, 2, 1>(sx, s_wT, bb3,  op, lane);
  tconv_branch<2, 3, 3, 2>(sx, s_wT, bb6,  op, lane);
  tconv_branch<6, 6, 6, 3>(sx, s_wT, bb12, op, lane);
}

// ---------------- K2: GCN linear xl = x . w_gcn^T (WMMA, K pad 16->32) ----------
__global__ __launch_bounds__(256) void k_xl(const float* __restrict__ x,
                                            const float* __restrict__ wg,
                                            __bf16* __restrict__ xl) {
  __shared__ __align__(16) __bf16 s_wg[16 * 32];   // [oc][c], cols 16..31 zero
  const int tid = threadIdx.x, lane = tid & 31, wv = tid >> 5;
  for (int i = tid; i < 16 * 32; i += 256) {
    int oc = i >> 5, c = i & 31;
    s_wg[i] = (c < 16) ? (__bf16)wg[oc * 16 + c] : (__bf16)0.f;
  }
  __syncthreads();

  const int oc = lane & 15, half = lane >> 4;
  const __bf16* pb = s_wg + oc * 32 + (half ? 8 : 0);
  const v16bf bm = ld_frag(pb, pb + 16);

  const long gw = (long)blockIdx.x * 8 + wv;       // over B*T*(N/64) = 8192
  const int ntg = (int)(gw & 15);                  // group of 4 n-tiles
  const long bt = gw >> 4;                         // b*T + t
  const int t = (int)(bt & 63), b = (int)(bt >> 4 >> 2);

  __bf16* xo = xl + bt * (NN * OCC);
#pragma unroll
  for (int nt = 0; nt < 4; ++nt) {
    const int ntile = ntg * 4 + nt;
    const int n = ntile * 16 + oc;                 // M row = n
    const float* px = x + (((long)b * NN + n) * TT + t) * CC + (half ? 8 : 0);
    union { v16bf v; } ua;
    const float4 r0 = *(const float4*)px;
    const float4 r1 = *(const float4*)(px + 4);
    ua.v[0] = (__bf16)r0.x; ua.v[1] = (__bf16)r0.y;
    ua.v[2] = (__bf16)r0.z; ua.v[3] = (__bf16)r0.w;
    ua.v[4] = (__bf16)r1.x; ua.v[5] = (__bf16)r1.y;
    ua.v[6] = (__bf16)r1.z; ua.v[7] = (__bf16)r1.w;
#pragma unroll
    for (int e = 8; e < 16; ++e) ua.v[e] = (__bf16)0.f;  // K=16..31 pad

    v8f acc = {};
    acc = wmma_bf16(ua.v, bm, acc);
#pragma unroll
    for (int r = 0; r < 8; ++r) {
      int nl = ntile * 16 + r + (half ? 8 : 0);
      xo[nl * OCC + oc] = (__bf16)acc[r];
    }
  }
}

// ---------------- K3: edge scatter in LDS (ds_add_f32), self-loop+bias fused ----
__global__ __launch_bounds__(1024) void k_scatter(
    const int* __restrict__ ei, const float* __restrict__ dinv,
    const float* __restrict__ bg, const __bf16* __restrict__ xl,
    __bf16* __restrict__ agg) {
  __shared__ float s_agg[NN * OCC];                // 64 KB (CDNA5: 320KB/WGP)
  const int tid = threadIdx.x;
  const long bt = blockIdx.x;                      // b*T + t
  const __bf16* xlp = xl + bt * (NN * OCC);

  for (int i = tid; i < NN * OCC; i += 1024) {     // init: dinv^2 * xl + bias
    int n = i >> 4, oc = i & 15;
    float di = dinv[n];
    s_agg[i] = di * di * (float)xlp[i] + bg[oc];
  }
  __syncthreads();

  const int* rowp = ei;
  const int* colp = ei + EE;
  for (int idx = tid; idx < EE * 2; idx += 1024) {
    int e = idx >> 1, h = (idx & 1) * 8;
    int r = rowp[e], c = colp[e];
    float nrm = dinv[r] * dinv[c];
    union { B16 q; __bf16 h8[8]; } u;
    u.q = *(const B16*)(xlp + r * OCC + h);        // 16B vector read
    float* dst = s_agg + c * OCC + h;
#pragma unroll
    for (int j = 0; j < 8; ++j) atomicAdd(&dst[j], nrm * (float)u.h8[j]);
  }
  __syncthreads();

  __bf16* ao = agg + bt * (NN * OCC);              // layout [B,T,N,OC]
  for (int i = tid; i < NN * OCC; i += 1024) ao[i] = (__bf16)s_agg[i];
}

// ---------------- K4: slin [16 x 64] x [64 x 64] per (b,n) (WMMA) + tanh -------
__global__ __launch_bounds__(256) void k_slin(
    const __bf16* __restrict__ agg, const float* __restrict__ wsl,
    const float* __restrict__ bsl, float* __restrict__ out) {
  __shared__ __align__(16) __bf16 s_w[64 * 64];    // w_slin[s][t'] as bf16
  const int tid = threadIdx.x, lane = tid & 31, wv = tid >> 5;
  for (int i = tid; i < 64 * 64; i += 256) s_w[i] = (__bf16)wsl[i];
  __syncthreads();

  const long gw = (long)blockIdx.x * 8 + wv;       // over B*N = 8192
  const int n = (int)(gw & 1023), b = (int)(gw >> 10);
  const int oc = lane & 15, half = lane >> 4;

  // B = agg[b, t', n, :]  (K=t', N=oc); two K-chunks. Scalar gathers: each
  // unrolled instruction reads two contiguous 32B runs across the wave.
  v16bf bf0, bf1;
  const __bf16* ap = agg + (long)b * TT * NN * OCC + n * 16 + oc;
#pragma unroll
  for (int e = 0; e < 16; ++e) {
    int kb = (e & 7) + (e >> 3) * 16 + (half ? 8 : 0);
    bf0[e] = ap[(long)kb * (NN * OCC)];
    bf1[e] = ap[(long)(kb + 32) * (NN * OCC)];
  }

  float* op = out + gw * (TT * 80) + 64;
#pragma unroll
  for (int m = 0; m < 4; ++m) {                    // 4 tiles of output s
    v8f acc = {};
#pragma unroll
    for (int ch = 0; ch < 2; ++ch) {
      const __bf16* pa0 = s_w + (m * 16 + oc) * 64 + ch * 32 + (half ? 8 : 0);
      const __bf16* pa1 = pa0 + 16;                // second 8-run (K+16)
      v16bf a = ld_frag(pa0, pa1);
      acc = wmma_bf16(a, (ch == 0) ? bf0 : bf1, acc);
    }
#pragma unroll
    for (int r = 0; r < 8; ++r) {
      int s = m * 16 + r + (half ? 8 : 0);
      op[(long)s * 80 + oc] = fast_tanh(acc[r] + bsl[s]);
    }
  }
}

// ---------------- launch ----------------
extern "C" void kernel_launch(void* const* d_in, const int* in_sizes, int n_in,
                              void* d_out, int out_size, void* d_ws, size_t ws_size,
                              hipStream_t stream) {
  (void)in_sizes; (void)n_in; (void)out_size; (void)ws_size;
  const float* x    = (const float*)d_in[0];
  const int*   ei   = (const int*)  d_in[1];
  const float* w2   = (const float*)d_in[2];
  const float* b2   = (const float*)d_in[3];
  const float* w3   = (const float*)d_in[4];
  const float* b3   = (const float*)d_in[5];
  const float* w6   = (const float*)d_in[6];
  const float* b6   = (const float*)d_in[7];
  const float* w12  = (const float*)d_in[8];
  const float* b12  = (const float*)d_in[9];
  const float* wg   = (const float*)d_in[10];
  const float* bg   = (const float*)d_in[11];
  const float* wsl  = (const float*)d_in[12];
  const float* bsl  = (const float*)d_in[13];
  float* out = (float*)d_out;

  char* ws = (char*)d_ws;
  float*  dinv = (float*)ws;                                    // 4 KB
  __bf16* xl   = (__bf16*)(ws + 4096);                          // 16.8 MB
  __bf16* agg  = (__bf16*)(ws + 4096 + (size_t)BB*TT*NN*OCC*2); // 16.8 MB

  k_deg    <<<1,    1024, 0, stream>>>(ei, dinv);
  k_tconv  <<<1024, 256,  0, stream>>>(x, w2, b2, w3, b3, w6, b6, w12, b12, out);
  k_xl     <<<1024, 256,  0, stream>>>(x, wg, xl);
  k_scatter<<<512,  1024, 0, stream>>>(ei, dinv, bg, xl, agg);
  k_slin   <<<1024, 256,  0, stream>>>(agg, wsl, bsl, out);
}
